// Features_18691697672212
// MI455X (gfx1250) — matrix operand order repaired and checked by reading, not compiled, not run on hardware
//
#include <hip/hip_runtime.h>
#include <hip/hip_bf16.h>
#include <math.h>

// ---------------------------------------------------------------------------
// Fused adaptive-kNN memory-matching for MI455X (gfx1250, wave32, WMMA + TDM).
//   dist^2 = |q|^2 + |m|^2 - 2 q.m  via bf16 WMMA (f32 accumulate)
//   A/B tiles staged global->LDS by the Tensor Data Mover (double-buffered,
//   TENSORcnt-tracked), streaming per-row top-5, small epilogue.
// ---------------------------------------------------------------------------

typedef __bf16 bf16x4  __attribute__((ext_vector_type(4)));
typedef __bf16 bf16x8  __attribute__((ext_vector_type(8)));
typedef __bf16 bf16x16 __attribute__((ext_vector_type(16)));
typedef float  f32x8   __attribute__((ext_vector_type(8)));
typedef unsigned int       u32;
typedef unsigned long long u64;
typedef u32 v4u __attribute__((ext_vector_type(4)));
typedef u32 v8u __attribute__((ext_vector_type(8)));

#define NQ 8192
#define NM 16384
#define DIM 1024
#define KTOP 5
#define NSPLIT 8
#define COLS_PER_SPLIT (NM / NSPLIT)          // 2048
#define CAND_PER_ROW (NSPLIT * 2 * KTOP)      // 80

// ---- TDM-path tiling (grid 128 x 8, 128 threads = 4 waves) ----
#define TM 64
#define TN 64
#define TK 64
#define LDA 72          // 64 bf16 data + 8 bf16 (16B) pad per row
#define LDS_SC 68
#define NCHUNK (COLS_PER_SPLIT / TN)          // 32
#define NKC (DIM / TK)                        // 16

// ---- Fallback-path tiling (grid 64 x 8, 256 threads = 8 waves) ----
#define F_TM 128
#define F_TN 64

// ---- Workspace layouts ----
// TDM path (bytes): qb16[16MB] | mb16[32MB] | x2 | y2 | pd | pi
#define WSB_QB16 0ull
#define WSB_MB16 (WSB_QB16 + (u64)NQ * DIM * 2)          // 16,777,216
#define WSB_X2   (WSB_MB16 + (u64)NM * DIM * 2)          // 50,331,648
#define WSB_Y2   (WSB_X2 + (u64)NQ * 4)
#define WSB_PD   (WSB_Y2 + (u64)NM * 4)
#define WSB_PI   (WSB_PD + (u64)NQ * CAND_PER_ROW * 4)
#define WSB_TOTAL (WSB_PI + (u64)NQ * CAND_PER_ROW * 4)  // ~53.1 MB
// Fallback path (floats): x2 | y2 | pd | pi  (~5.3 MB)
#define FWS_X2 0
#define FWS_Y2 (FWS_X2 + NQ)
#define FWS_PD (FWS_Y2 + NM)
#define FWS_PI (FWS_PD + NQ * CAND_PER_ROW)
#define FWS_TOTAL_BYTES ((u64)(FWS_PI + NQ * CAND_PER_ROW) * 4)

// ---------------------------------------------------------------------------
__device__ __forceinline__ void top5_insert(float d, int idx, float* s, int* ii) {
    if (d < s[4]) {
        s[4] = d; ii[4] = idx;
#pragma unroll
        for (int j = 4; j > 0; --j) {
            if (s[j] < s[j - 1]) {
                float td = s[j]; s[j] = s[j - 1]; s[j - 1] = td;
                int ti = ii[j]; ii[j] = ii[j - 1]; ii[j - 1] = ti;
            }
        }
    }
}

// ---------------------------------------------------------------------------
// per-row squared L2 norm, one wave per row
// ---------------------------------------------------------------------------
__global__ __launch_bounds__(256) void rownorm2_kernel(const float* __restrict__ x,
                                                       float* __restrict__ out,
                                                       int nrows) {
    int row  = blockIdx.x * (blockDim.x >> 5) + (threadIdx.x >> 5);
    int lane = threadIdx.x & 31;
    if (row >= nrows) return;
    const float* p = x + (size_t)row * DIM;
    float s = 0.0f;
    for (int c = lane * 4; c < DIM; c += 32 * 4) {
        float4 v = *(const float4*)(p + c);
        s += v.x * v.x + v.y * v.y + v.z * v.z + v.w * v.w;
    }
#pragma unroll
    for (int off = 16; off > 0; off >>= 1) s += __shfl_xor(s, off, 32);
    if (lane == 0) out[row] = s;
}

// ---------------------------------------------------------------------------
// one-time f32 -> bf16 mirror (row-major, unpadded) for the TDM path
// ---------------------------------------------------------------------------
__global__ __launch_bounds__(256) void conv_bf16_kernel(const float* __restrict__ src,
                                                        __bf16* __restrict__ dst,
                                                        int n4) {
    int idx = blockIdx.x * blockDim.x + threadIdx.x;
    int stride = gridDim.x * blockDim.x;
    for (; idx < n4; idx += stride) {
        float4 v = ((const float4*)src)[idx];
        bf16x4 b = { (__bf16)v.x, (__bf16)v.y, (__bf16)v.z, (__bf16)v.w };
        ((bf16x4*)dst)[idx] = b;
    }
}

// ---------------------------------------------------------------------------
// WMMA fragment loads from padded row-major LDS tiles (CDNA5 layouts)
// ---------------------------------------------------------------------------
__device__ __forceinline__ bf16x16 load_a_frag(const __bf16* tile, int ks, int lane) {
    int r = lane & 15, h = lane >> 4;
    const __bf16* p = tile + r * LDA + ks + h * 8;
    bf16x8 lo = *(const bf16x8*)(p);
    bf16x8 hi = *(const bf16x8*)(p + 16);
    return __builtin_shufflevector(lo, hi, 0,1,2,3,4,5,6,7,8,9,10,11,12,13,14,15);
}
__device__ __forceinline__ bf16x16 load_b_frag(const __bf16* tile, int ks, int lane) {
    int r = lane & 15, h = lane >> 4;
    const __bf16* p = tile + r * LDA + ks + h * 8;
    bf16x8 lo = *(const bf16x8*)(p);
    bf16x8 hi = *(const bf16x8*)(p + 16);
    return __builtin_shufflevector(lo, hi, 0,1,2,3,4,5,6,7,8,9,10,11,12,13,14,15);
}

// ---------------------------------------------------------------------------
// TDM 2-D tile load: global (bf16, row-major) -> LDS with 16B row padding.
// D# groups per CDNA5 ISA ss8.3/8.4; 2-D => groups 2/3 NULL.
//   data_size=1 (2B), pad_interval=4 (pad every 128B), pad_amount=3 (4 dwords)
// ---------------------------------------------------------------------------
__device__ __forceinline__ void tdm_load_2d(u32 lds_off, u64 gaddr,
                                            u32 tensor_d0, u32 tensor_d1,
                                            u32 tile_d0, u32 tile_d1,
                                            u32 stride0) {
    v4u g0;
    g0[0] = 1u;                                          // count=1, user desc
    g0[1] = lds_off;                                     // LDS byte address
    g0[2] = (u32)(gaddr & 0xFFFFFFFFull);                // global addr lo
    g0[3] = (u32)((gaddr >> 32) & 0x01FFFFFFull) | (2u << 30);  // hi + type=2

    const u32 pad_interval = 4u;   // 8<<4 = 128 bytes between pads
    const u32 pad_amount   = 3u;   // (3+1) dwords = 16 bytes of pad
    u64 q0 = (u64)((1u << 16)                      // data_size = 2 bytes
                 | (1u << 20)                      // pad_enable
                 | (pad_interval << 22)
                 | (pad_amount << 25))
           | ((u64)(tensor_d0 & 0xFFFFu) << 48);   // tensor_dim0 [79:48] lo
    u64 q1 = (u64)(tensor_d0 >> 16)                // tensor_dim0 hi
           | ((u64)tensor_d1 << 16)                // tensor_dim1 [111:80]
           | ((u64)tile_d0 << 48);                 // tile_dim0 [127:112]
    u64 q2 = (u64)tile_d1                          // tile_dim1 [143:128]
           | ((u64)stride0 << 32);                 // tensor_dim0_stride lo32
    u64 q3 = 0ull;                                 // stride0 hi16=0, stride1=0

    v8u g1;
    g1[0] = (u32)q0; g1[1] = (u32)(q0 >> 32);
    g1[2] = (u32)q1; g1[3] = (u32)(q1 >> 32);
    g1[4] = (u32)q2; g1[5] = (u32)(q2 >> 32);
    g1[6] = (u32)q3; g1[7] = (u32)(q3 >> 32);

    asm volatile("tensor_load_to_lds %0, %1" :: "s"(g0), "s"(g1) : "memory");
}

// ---------------------------------------------------------------------------
// TDM main kernel: grid (NQ/TM=128, NSPLIT=8), 128 threads = 4 waves (2x2).
// ---------------------------------------------------------------------------
__global__ __launch_bounds__(128) void knn_topk_gemm_tdm(const __bf16* __restrict__ qb,
                                                         const __bf16* __restrict__ mb16,
                                                         const float* __restrict__ x2,
                                                         const float* __restrict__ y2,
                                                         float* __restrict__ pd,
                                                         int* __restrict__ pi) {
    __shared__ __align__(16) __bf16 As[2][TM * LDA];   // 2 x  9,216 B
    __shared__ __align__(16) __bf16 Bs[2][TN * LDA];   // 2 x  9,216 B
    __shared__ __align__(16) float  Sc[TM * LDS_SC];   //     17,408 B (54,272 total)

    const int t    = threadIdx.x;
    const int lane = t & 31;
    const int wave = t >> 5;
    const int wm   = wave & 1;   // 2 waves along M (32 rows each)
    const int wn   = wave >> 1;  // 2 waves along N (32 cols each)
    const int rowBase   = blockIdx.x * TM;
    const int splitBase = blockIdx.y * COLS_PER_SPLIT;

    const int sr = t & 63;       // scan row
    const int sh = t >> 6;       // scan half (2 threads per row, 32 cols each)
    const float x2r = x2[rowBase + sr];

    const u32 asOff0 = (u32)(uintptr_t)&As[0][0];
    const u32 asOff1 = (u32)(uintptr_t)&As[1][0];
    const u32 bsOff0 = (u32)(uintptr_t)&Bs[0][0];
    const u32 bsOff1 = (u32)(uintptr_t)&Bs[1][0];

    float s[KTOP]; int ii[KTOP];
#pragma unroll
    for (int j = 0; j < KTOP; ++j) { s[j] = 3.0e38f; ii[j] = 0; }

    for (int nc = 0; nc < NCHUNK; ++nc) {
        const int colBase = splitBase + nc * TN;
        const u64 aRowByte = 2ull * ((u64)rowBase * DIM);
        const u64 bRowByte = 2ull * ((u64)colBase * DIM);

        f32x8 acc00 = {}; f32x8 acc01 = {}; f32x8 acc10 = {}; f32x8 acc11 = {};

        // prologue: prime both buffers (4 tensor ops in flight on wave 0)
        if (wave == 0) {
            tdm_load_2d(asOff0, (u64)(uintptr_t)qb   + aRowByte + 2ull * 0,
                        DIM, NQ, TK, TM, DIM);
            tdm_load_2d(bsOff0, (u64)(uintptr_t)mb16 + bRowByte + 2ull * 0,
                        DIM, NM, TK, TN, DIM);
            tdm_load_2d(asOff1, (u64)(uintptr_t)qb   + aRowByte + 2ull * TK,
                        DIM, NQ, TK, TM, DIM);
            tdm_load_2d(bsOff1, (u64)(uintptr_t)mb16 + bRowByte + 2ull * TK,
                        DIM, NM, TK, TN, DIM);
        }

        for (int kc = 0; kc < NKC; ++kc) {
            const int buf = kc & 1;
            if (wave == 0) {
                if (kc < NKC - 1) __builtin_amdgcn_s_wait_tensorcnt(2);
                else              __builtin_amdgcn_s_wait_tensorcnt(0);
            }
            __syncthreads();   // buf's tiles are resident for all waves

#pragma unroll
            for (int ks = 0; ks < TK; ks += 32) {
                bf16x16 a0 = load_a_frag(&As[buf][(wm * 32 + 0)  * LDA], ks, lane);
                bf16x16 a1 = load_a_frag(&As[buf][(wm * 32 + 16) * LDA], ks, lane);
                bf16x16 b0 = load_b_frag(&Bs[buf][(wn * 32 + 0)  * LDA], ks, lane);
                bf16x16 b1 = load_b_frag(&Bs[buf][(wn * 32 + 16) * LDA], ks, lane);
                acc00 = __builtin_amdgcn_wmma_f32_16x16x32_bf16(false, a0, false, b0, (short)0, acc00, false, false);
                acc01 = __builtin_amdgcn_wmma_f32_16x16x32_bf16(false, a0, false, b1, (short)0, acc01, false, false);
                acc10 = __builtin_amdgcn_wmma_f32_16x16x32_bf16(false, a1, false, b0, (short)0, acc10, false, false);
                acc11 = __builtin_amdgcn_wmma_f32_16x16x32_bf16(false, a1, false, b1, (short)0, acc11, false, false);
            }
            __syncthreads();   // everyone done reading buf before DMA reuses it

            if (wave == 0 && kc + 2 < NKC) {
                const u64 kByte = 2ull * (u64)((kc + 2) * TK);
                tdm_load_2d(buf ? asOff1 : asOff0,
                            (u64)(uintptr_t)qb + aRowByte + kByte,
                            DIM, NQ, TK, TM, DIM);
                tdm_load_2d(buf ? bsOff1 : bsOff0,
                            (u64)(uintptr_t)mb16 + bRowByte + kByte,
                            DIM, NM, TK, TN, DIM);
            }
        }

        // scatter dot products into the padded score tile
        {
            const int cr = lane & 15, ch = lane >> 4;
#pragma unroll
            for (int mi = 0; mi < 2; ++mi)
#pragma unroll
                for (int ni = 0; ni < 2; ++ni) {
                    f32x8 a = (mi == 0) ? (ni == 0 ? acc00 : acc01)
                                        : (ni == 0 ? acc10 : acc11);
                    float* dst = Sc + (wm * 32 + mi * 16 + ch * 8) * LDS_SC + wn * 32 + ni * 16 + cr;
#pragma unroll
                    for (int v = 0; v < 8; ++v) dst[v * LDS_SC] = a[v];
                }
        }
        __syncthreads();

        // fused top-5 scan on d^2 = |q|^2 + |m|^2 - 2 dot
        {
            const float* srow = Sc + sr * LDS_SC + sh * 32;
            const int cb = colBase + sh * 32;
#pragma unroll 4
            for (int c = 0; c < 32; ++c) {
                float d2 = x2r + y2[cb + c] - 2.0f * srow[c];
                top5_insert(d2, cb + c, s, ii);
            }
        }
        __syncthreads();
    }

    size_t base = ((size_t)(rowBase + sr) * NSPLIT + blockIdx.y) * (2 * KTOP) + sh * KTOP;
#pragma unroll
    for (int j = 0; j < KTOP; ++j) { pd[base + j] = s[j]; pi[base + j] = ii[j]; }
}

// ---------------------------------------------------------------------------
// Fallback kernel (no ws room for bf16 mirrors): in-kernel f32->bf16 staging.
// grid (NQ/F_TM=64, NSPLIT), 256 threads = 8 waves (4x2).
// ---------------------------------------------------------------------------
__global__ __launch_bounds__(256) void knn_topk_gemm_fb(const float* __restrict__ q,
                                                        const float* __restrict__ mb,
                                                        const float* __restrict__ x2,
                                                        const float* __restrict__ y2,
                                                        float* __restrict__ pd,
                                                        int* __restrict__ pi) {
    __shared__ __align__(16) __bf16 As[F_TM * LDA];
    __shared__ __align__(16) __bf16 Bs[F_TN * LDA];
    __shared__ __align__(16) float  Sc[F_TM * LDS_SC];

    const int t    = threadIdx.x;
    const int lane = t & 31;
    const int wave = t >> 5;
    const int wm   = wave & 3;
    const int wn   = wave >> 2;
    const int rowBase   = blockIdx.x * F_TM;
    const int splitBase = blockIdx.y * COLS_PER_SPLIT;

    const int sr = t & 127;
    const int sh = t >> 7;
    const float x2r = x2[rowBase + sr];

    float s[KTOP]; int ii[KTOP];
#pragma unroll
    for (int j = 0; j < KTOP; ++j) { s[j] = 3.0e38f; ii[j] = 0; }

    for (int nc = 0; nc < COLS_PER_SPLIT / F_TN; ++nc) {
        const int colBase = splitBase + nc * F_TN;
        f32x8 acc00 = {}; f32x8 acc01 = {}; f32x8 acc10 = {}; f32x8 acc11 = {};

        for (int kc = 0; kc < DIM / TK; ++kc) {
            const int kBase = kc * TK;
            __syncthreads();
#pragma unroll
            for (int i = 0; i < 8; ++i) {
                int idx = t + i * 256;
                int r = idx >> 4, cv = idx & 15;
                float4 v = *(const float4*)(q + (size_t)(rowBase + r) * DIM + kBase + cv * 4);
                bf16x4 b = { (__bf16)v.x, (__bf16)v.y, (__bf16)v.z, (__bf16)v.w };
                *(bf16x4*)(As + r * LDA + cv * 4) = b;
            }
#pragma unroll
            for (int i = 0; i < 4; ++i) {
                int idx = t + i * 256;
                int r = idx >> 4, cv = idx & 15;
                float4 v = *(const float4*)(mb + (size_t)(colBase + r) * DIM + kBase + cv * 4);
                bf16x4 b = { (__bf16)v.x, (__bf16)v.y, (__bf16)v.z, (__bf16)v.w };
                *(bf16x4*)(Bs + r * LDA + cv * 4) = b;
            }
            __syncthreads();
#pragma unroll
            for (int ks = 0; ks < TK; ks += 32) {
                bf16x16 a0 = load_a_frag(As + (wm * 32 + 0)  * LDA, ks, lane);
                bf16x16 a1 = load_a_frag(As + (wm * 32 + 16) * LDA, ks, lane);
                bf16x16 b0 = load_b_frag(Bs + (wn * 32 + 0)  * LDA, ks, lane);
                bf16x16 b1 = load_b_frag(Bs + (wn * 32 + 16) * LDA, ks, lane);
                acc00 = __builtin_amdgcn_wmma_f32_16x16x32_bf16(false, a0, false, b0, (short)0, acc00, false, false);
                acc01 = __builtin_amdgcn_wmma_f32_16x16x32_bf16(false, a0, false, b1, (short)0, acc01, false, false);
                acc10 = __builtin_amdgcn_wmma_f32_16x16x32_bf16(false, a1, false, b0, (short)0, acc10, false, false);
                acc11 = __builtin_amdgcn_wmma_f32_16x16x32_bf16(false, a1, false, b1, (short)0, acc11, false, false);
            }
        }
        {
            const int cr = lane & 15, ch = lane >> 4;
#pragma unroll
            for (int mi = 0; mi < 2; ++mi)
#pragma unroll
                for (int ni = 0; ni < 2; ++ni) {
                    f32x8 a = (mi == 0) ? (ni == 0 ? acc00 : acc01)
                                        : (ni == 0 ? acc10 : acc11);
                    float* dst = Sc + (wm * 32 + mi * 16 + ch * 8) * LDS_SC + wn * 32 + ni * 16 + cr;
#pragma unroll
                    for (int v = 0; v < 8; ++v) dst[v * LDS_SC] = a[v];
                }
        }
        __syncthreads();
        {
            const float* srow = Sc + sr * LDS_SC + sh * 32;
            const int cb = colBase + sh * 32;
#pragma unroll 4
            for (int c = 0; c < 32; ++c) {
                float d2 = x2r + y2[cb + c] - 2.0f * srow[c];
                top5_insert(d2, cb + c, s, ii);
            }
        }
        __syncthreads();
    }

    size_t base = ((size_t)(rowBase + sr) * NSPLIT + blockIdx.y) * (2 * KTOP) + sh * KTOP;
#pragma unroll
    for (int j = 0; j < KTOP; ++j) { pd[base + j] = s[j]; pi[base + j] = ii[j]; }
}

// ---------------------------------------------------------------------------
// Finalize: merge 80 candidates/row, softmax-weighted distance, consistency.
// ---------------------------------------------------------------------------
__global__ __launch_bounds__(256) void finalize_kernel(const float* __restrict__ pd,
                                                       const int* __restrict__ pi,
                                                       const float* __restrict__ nsb,
                                                       float* __restrict__ out) {
    int row = blockIdx.x * blockDim.x + threadIdx.x;
    if (row >= NQ) return;

    float s[KTOP]; int ii[KTOP];
#pragma unroll
    for (int j = 0; j < KTOP; ++j) { s[j] = 3.0e38f; ii[j] = 0; }

    size_t base = (size_t)row * CAND_PER_ROW;
    for (int j = 0; j < CAND_PER_ROW; ++j)
        top5_insert(pd[base + j], pi[base + j], s, ii);

    float d[KTOP];
#pragma unroll
    for (int j = 0; j < KTOP; ++j) d[j] = sqrtf(fmaxf(s[j], 1e-12f));

    float wsum = 0.0f, wd = 0.0f;
#pragma unroll
    for (int j = 0; j < KTOP; ++j) { float w = __expf(d[0] - d[j]); wsum += w; wd += w * d[j]; }
    wd /= wsum;

    float scale = 0.0f;
#pragma unroll
    for (int j = 0; j < KTOP; ++j) scale += nsb[ii[j]];
    scale *= 0.2f;
    float nd = wd / fmaxf(scale, 1e-6f);

    float mean = 0.0f;
#pragma unroll
    for (int j = 0; j < KTOP; ++j) mean += d[j];
    mean *= 0.2f;
    float var = 0.0f;
#pragma unroll
    for (int j = 0; j < KTOP; ++j) { float e = d[j] - mean; var += e * e; }
    var *= 0.2f;
    float cons = sqrtf(var) / fmaxf(mean, 1e-6f);

    out[row] = nd * (1.0f + 0.5f * cons);
}

// ---------------------------------------------------------------------------
extern "C" void kernel_launch(void* const* d_in, const int* in_sizes, int n_in,
                              void* d_out, int out_size, void* d_ws, size_t ws_size,
                              hipStream_t stream) {
    (void)in_sizes; (void)n_in; (void)out_size;
    const float* q   = (const float*)d_in[0];
    const float* mb  = (const float*)d_in[1];
    const float* nsb = (const float*)d_in[2];
    char* wsb = (char*)d_ws;

    if (ws_size >= WSB_TOTAL) {
        // TDM path: bf16 mirrors + descriptor-driven DMA staging
        __bf16* qb16 = (__bf16*)(wsb + WSB_QB16);
        __bf16* mb16 = (__bf16*)(wsb + WSB_MB16);
        float*  x2   = (float*)(wsb + WSB_X2);
        float*  y2   = (float*)(wsb + WSB_Y2);
        float*  pd   = (float*)(wsb + WSB_PD);
        int*    pi   = (int*)(wsb + WSB_PI);

        conv_bf16_kernel<<<2048, 256, 0, stream>>>(q,  qb16, NQ * DIM / 4);
        conv_bf16_kernel<<<2048, 256, 0, stream>>>(mb, mb16, NM * DIM / 4);
        rownorm2_kernel<<<NQ / 8, 256, 0, stream>>>(q,  x2, NQ);
        rownorm2_kernel<<<NM / 8, 256, 0, stream>>>(mb, y2, NM);

        dim3 grid(NQ / TM, NSPLIT);
        knn_topk_gemm_tdm<<<grid, 128, 0, stream>>>(qb16, mb16, x2, y2, pd, pi);
        finalize_kernel<<<(NQ + 255) / 256, 256, 0, stream>>>(pd, pi, nsb, (float*)d_out);
    } else {
        // fallback: fused f32->bf16 staging inside the GEMM kernel
        float* wsf = (float*)d_ws;
        float* x2 = wsf + FWS_X2;
        float* y2 = wsf + FWS_Y2;
        float* pd = wsf + FWS_PD;
        int*   pi = (int*)(wsf + FWS_PI);

        rownorm2_kernel<<<NQ / 8, 256, 0, stream>>>(q,  x2, NQ);
        rownorm2_kernel<<<NM / 8, 256, 0, stream>>>(mb, y2, NM);

        dim3 grid(NQ / F_TM, NSPLIT);
        knn_topk_gemm_fb<<<grid, 256, 0, stream>>>(q, mb, x2, y2, pd, pi);
        finalize_kernel<<<(NQ + 255) / 256, 256, 0, stream>>>(pd, pi, nsb, (float*)d_out);
    }
}
